// PAM_Module_20160576487497
// MI455X (gfx1250) — matrix operand order repaired and checked by reading, not compiled, not run on hardware
//
#include <hip/hip_runtime.h>

// Problem constants: B=2, S=64, K=64 -> N=4096 tokens, D=256, C=32
#define NTOK 4096
#define DDIM 256
#define CDIM 32
#define PSTR 72   // padded P row stride (ushorts)
#define VSTR 72   // padded V-tile row stride (64 keys + 8 pad)
#define KSTR 40   // padded K-tile row stride (32 c + 8 pad)

typedef __attribute__((ext_vector_type(16))) __bf16 bf16x16;
typedef __attribute__((ext_vector_type(8)))  float  f32x8;
typedef __attribute__((ext_vector_type(4)))  unsigned u32x4;   // builtin vec: AS-safe

union Frag { u32x4 q[2]; bf16x16 v; };

__device__ __forceinline__ unsigned short f2bf(float f) {
  union { float f; unsigned u; } c; c.f = f;
  unsigned r = c.u + 0x7FFFu + ((c.u >> 16) & 1u);   // round-to-nearest-even
  return (unsigned short)(r >> 16);
}

// ---- forced-LDS (addrspace 3) accessors: low 32 bits of a generic LDS address
// ---- are the LDS byte offset, so inttoptr to AS3 yields true ds_* ops.
__device__ __forceinline__ u32x4 lds_ld16(const unsigned short* p) {
  const __attribute__((address_space(3))) u32x4* lp =
      (const __attribute__((address_space(3))) u32x4*)(size_t)(unsigned)(size_t)p;
  return *lp;
}
__device__ __forceinline__ void lds_st2(unsigned short* p, unsigned short v) {
  __attribute__((address_space(3))) unsigned short* lp =
      (__attribute__((address_space(3))) unsigned short*)(size_t)(unsigned)(size_t)p;
  *lp = v;
}

// A fragment 16x32 bf16 (MxK), source row-major [16][stride].
__device__ __forceinline__ bf16x16 load_a(const unsigned short* base, int stride, int lane) {
  const unsigned short* p = base + (lane & 15) * stride;
  int h = lane >> 4;
  Frag f;
  f.q[0] = *(const u32x4*)(p + h * 8);
  f.q[1] = *(const u32x4*)(p + 16 + h * 8);
  return f.v;
}

// B fragment 32x16 bf16 (KxN), source stored as [N][K] row-major (K contiguous per column).
__device__ __forceinline__ bf16x16 load_b(const unsigned short* base, int stride, int lane) {
  const unsigned short* p = base + (lane & 15) * stride + (lane >> 4) * 8;
  Frag f;
  f.q[0] = *(const u32x4*)(p);
  f.q[1] = *(const u32x4*)(p + 16);
  return f.v;
}

// LDS variants (guaranteed ds_load_b128)
__device__ __forceinline__ bf16x16 load_a_lds(const unsigned short* base, int stride, int lane) {
  const unsigned short* p = base + (lane & 15) * stride;
  int h = lane >> 4;
  Frag f;
  f.q[0] = lds_ld16(p + h * 8);
  f.q[1] = lds_ld16(p + 16 + h * 8);
  return f.v;
}
__device__ __forceinline__ bf16x16 load_b_lds(const unsigned short* base, int stride, int lane) {
  const unsigned short* p = base + (lane & 15) * stride + (lane >> 4) * 8;
  Frag f;
  f.q[0] = lds_ld16(p);
  f.q[1] = lds_ld16(p + 16);
  return f.v;
}

__device__ __forceinline__ f32x8 wmma_bf16(bf16x16 a, bf16x16 b, f32x8 c) {
  return __builtin_amdgcn_wmma_f32_16x16x32_bf16(false, a, false, b, (short)0, c, false, false);
}

// 16B async DMA: global -> LDS, tracked by ASYNCcnt (no VGPR return).
__device__ __forceinline__ void async_cp16(const unsigned short* lds_dst, const void* gsrc) {
  unsigned lds_off = (unsigned)(size_t)lds_dst;
  asm volatile("global_load_async_to_lds_b128 %0, %1, off"
               :: "v"(lds_off), "v"(gsrc) : "memory");
}

__device__ __forceinline__ void wait_async0() {
#if __has_builtin(__builtin_amdgcn_s_wait_asynccnt)
  __builtin_amdgcn_s_wait_asynccnt(0);
#else
  asm volatile("s_wait_asynccnt 0x0" ::: "memory");
#endif
}

// ---------- conversion kernels ----------
__global__ __launch_bounds__(256) void k_convert_x(const float* __restrict__ x,
                                                   unsigned short* __restrict__ Xbf, int total) {
  int idx = blockIdx.x * blockDim.x + threadIdx.x;
  if (idx >= total) return;
  int d = idx & (DDIM - 1);
  int n = (idx >> 8) & (NTOK - 1);
  int b = idx >> 20;
  int s = n & 63, kk = n >> 6;                      // n = kk*64 + s
  size_t src = (((size_t)b * 64 + s) * 64 + kk) * DDIM + d;
  Xbf[idx] = f2bf(x[src]);
}

__global__ __launch_bounds__(256) void k_convert_w(const float* __restrict__ Wq,
                                                   const float* __restrict__ Wk,
                                                   const float* __restrict__ Wv,
                                                   unsigned short* __restrict__ Wqb,
                                                   unsigned short* __restrict__ Wkb,
                                                   unsigned short* __restrict__ Wvb) {
  int idx = blockIdx.x * blockDim.x + threadIdx.x;
  if (idx < 8192)        Wqb[idx]         = f2bf(Wq[idx]);
  else if (idx < 16384)  Wkb[idx - 8192]  = f2bf(Wk[idx - 8192]);
  else if (idx < 81920)  Wvb[idx - 16384] = f2bf(Wv[idx - 16384]);
}

// ---------- Q/K projection: [4096,256] x [256,32] per batch (1-deep reg pipeline) ----------
__global__ __launch_bounds__(256) void k_proj_qk(const unsigned short* __restrict__ Xbf,
                                                 const unsigned short* __restrict__ Wqb,
                                                 const unsigned short* __restrict__ Wkb,
                                                 const float* __restrict__ bq,
                                                 const float* __restrict__ bk,
                                                 unsigned short* __restrict__ Qbf,
                                                 unsigned short* __restrict__ Kbf) {
  const int b = blockIdx.y;
  const int w = threadIdx.x >> 5, lane = threadIdx.x & 31;
  const int h = lane >> 4, l16 = lane & 15;
  const int tb = blockIdx.x * 128 + w * 16;
  const unsigned short* Xrow = Xbf + ((size_t)b * NTOK + tb) * DDIM;
  f32x8 aq[2] = {}; f32x8 ak[2] = {};

  bf16x16 Acur = load_a(Xrow, DDIM, lane);
  bf16x16 Bn = load_b(Wqb, DDIM, lane);             // step 0: q, j=0, kc=0
  #pragma unroll
  for (int kc = 0; kc < 8; ++kc) {
    bf16x16 A = Acur;
    if (kc < 7) Acur = load_a(Xrow + (kc + 1) * 32, DDIM, lane);
    #pragma unroll
    for (int st = 0; st < 4; ++st) {                // st: 0=q/j0, 1=k/j0, 2=q/j1, 3=k/j1
      bf16x16 Bc = Bn;
      int g = kc * 4 + st + 1;
      if (g < 32) {
        int kcn = g >> 2, stn = g & 3;
        const unsigned short* W = (stn & 1) ? Wkb : Wqb;
        Bn = load_b(W + ((stn >> 1) * 16) * DDIM + kcn * 32, DDIM, lane);
      }
      if (st & 1) ak[st >> 1] = wmma_bf16(A, Bc, ak[st >> 1]);
      else        aq[st >> 1] = wmma_bf16(A, Bc, aq[st >> 1]);
    }
  }
  #pragma unroll
  for (int j = 0; j < 2; ++j) {
    int c = j * 16 + l16;
    float bqv = bq[c], bkv = bk[c];
    #pragma unroll
    for (int i = 0; i < 8; ++i) {
      int tok = tb + i + 8 * h;
      Qbf[((size_t)b * NTOK + tok) * CDIM + c] = f2bf(aq[j][i] + bqv);
      Kbf[((size_t)b * NTOK + tok) * CDIM + c] = f2bf(ak[j][i] + bkv);
    }
  }
}

// ---------- V projection producing Vt[d][n] (1-deep reg pipeline) ----------
__global__ __launch_bounds__(256) void k_proj_v(const unsigned short* __restrict__ Xbf,
                                                const unsigned short* __restrict__ Wvb,
                                                const float* __restrict__ bv,
                                                unsigned short* __restrict__ Vt) {
  const int b = blockIdx.z;
  const int w = threadIdx.x >> 5, lane = threadIdx.x & 31;
  const int h = lane >> 4, l16 = lane & 15;
  const int mw = w >> 1, nw = w & 1;
  const int db = blockIdx.y * 64 + mw * 16;
  const int nb = blockIdx.x * 256 + nw * 128;
  const unsigned short* XB = Xbf + ((size_t)b * NTOK + nb) * DDIM;
  f32x8 acc[8] = {};

  bf16x16 Acur = load_a(Wvb + db * DDIM, DDIM, lane);
  bf16x16 Bn = load_b(XB, DDIM, lane);              // u = 0: kc=0, j=0
  #pragma unroll
  for (int kc = 0; kc < 8; ++kc) {
    bf16x16 A = Acur;
    if (kc < 7) Acur = load_a(Wvb + db * DDIM + (kc + 1) * 32, DDIM, lane);
    #pragma unroll
    for (int j = 0; j < 8; ++j) {
      bf16x16 Bc = Bn;
      int u = kc * 8 + j + 1;
      if (u < 64)
        Bn = load_b(XB + (size_t)(u & 7) * 16 * DDIM + (u >> 3) * 32, DDIM, lane);
      acc[j] = wmma_bf16(A, Bc, acc[j]);
    }
  }
  #pragma unroll
  for (int j = 0; j < 8; ++j) {
    #pragma unroll
    for (int i = 0; i < 8; ++i) {
      int d = db + i + 8 * h;
      int n = nb + j * 16 + l16;
      Vt[((size_t)b * DDIM + d) * NTOK + n] = f2bf(acc[j][i] + bv[d]);
    }
  }
}

// Stage one 64-key block: K tile [64 x 32] and V tile [256 x 64] -> LDS (async DMA).
__device__ __forceinline__ void stage_tiles(const unsigned short* Kg,   // + (b*N + kb)*32
                                            const unsigned short* Vg,   // + b*D*N + kb
                                            unsigned short* Ks, unsigned short* Vs, int tid) {
  const unsigned short* vg = Vg + (size_t)tid * NTOK;
  unsigned short* vl = Vs + tid * VSTR;
  #pragma unroll
  for (int c = 0; c < 8; ++c) async_cp16(vl + c * 8, vg + c * 8);
  int kr = tid >> 2, kc = tid & 3;
  async_cp16(Ks + kr * KSTR + kc * 8, Kg + kr * 32 + kc * 8);
}

// ---------- flash attention: async double-buffered K/V tiles + fused epilogue ----------
__global__ __launch_bounds__(256, 1) void k_attn(const unsigned short* __restrict__ Qbf,
                                                 const unsigned short* __restrict__ Kbf,
                                                 const unsigned short* __restrict__ Vt,
                                                 const float* __restrict__ x,
                                                 const float* __restrict__ gamma,
                                                 float* __restrict__ out) {
  extern __shared__ __align__(16) unsigned short smem[];
  unsigned short* Vs0 = smem;                       // 2 x 256*VSTR
  unsigned short* Ks0 = Vs0 + 2 * 256 * VSTR;       // 2 x 64*KSTR
  unsigned short* Pst = Ks0 + 2 * 64 * KSTR;        // 64*PSTR

  const int b = blockIdx.y;
  const int qb = blockIdx.x * 64;
  const int tid = threadIdx.x;
  const int w = tid >> 5, lane = tid & 31;
  const int h = lane >> 4, l16 = lane & 15;
  const int mw = w >> 1;                            // query 16-row group (0..3)
  const int dw = w & 1;                             // D half (0..1), 128 each

  bf16x16 Aq = load_a(Qbf + ((size_t)b * NTOK + qb + mw * 16) * CDIM, CDIM, lane);

  f32x8 acc[8] = {};
  float m_run[8], l_run[8];
  #pragma unroll
  for (int i = 0; i < 8; ++i) { m_run[i] = -3.0e38f; l_run[i] = 0.0f; }

  const unsigned short* Kg0 = Kbf + (size_t)b * NTOK * CDIM;
  const unsigned short* Vg0 = Vt + (size_t)b * DDIM * NTOK;

  stage_tiles(Kg0, Vg0, Ks0, Vs0, tid);             // prologue DMA for block 0

  for (int it = 0; it < NTOK / 64; ++it) {
    const int kb = it * 64;
    const int buf = it & 1;
    unsigned short* Ks = Ks0 + buf * (64 * KSTR);   // offset select (no pointer phi)
    unsigned short* Vs = Vs0 + buf * (256 * VSTR);

    wait_async0();                                  // my DMA for this buffer done
    __syncthreads();                                // everyone's DMA done; prev reads retired
    if (it + 1 < NTOK / 64)                         // kick DMA for next block under compute
      stage_tiles(Kg0 + (size_t)(kb + 64) * CDIM, Vg0 + kb + 64,
                  Ks0 + (1 - buf) * (64 * KSTR), Vs0 + (1 - buf) * (256 * VSTR), tid);

    // S = Q K^T : 16 queries x 64 keys (C = 32 -> one WMMA per 16x16 tile)
    f32x8 s[4];
    #pragma unroll
    for (int j = 0; j < 4; ++j) {
      bf16x16 Bk = load_b_lds(Ks + (j * 16) * KSTR, KSTR, lane);
      f32x8 z = {};
      s[j] = wmma_bf16(Aq, Bk, z);
    }
    // online softmax: each row lives in 16 lanes of a half-wave -> xor-butterfly reduce
    float mnew[8], scale[8];
    #pragma unroll
    for (int i = 0; i < 8; ++i) {
      float v = fmaxf(fmaxf(s[0][i], s[1][i]), fmaxf(s[2][i], s[3][i]));
      v = fmaxf(v, __shfl_xor(v, 1));
      v = fmaxf(v, __shfl_xor(v, 2));
      v = fmaxf(v, __shfl_xor(v, 4));
      v = fmaxf(v, __shfl_xor(v, 8));
      mnew[i] = fmaxf(m_run[i], v);
      scale[i] = __expf(m_run[i] - mnew[i]);
      m_run[i] = mnew[i];
    }
    #pragma unroll
    for (int j = 0; j < 4; ++j)
      #pragma unroll
      for (int i = 0; i < 8; ++i)
        s[j][i] = __expf(s[j][i] - mnew[i]);
    #pragma unroll
    for (int i = 0; i < 8; ++i) {
      float r = s[0][i] + s[1][i] + s[2][i] + s[3][i];
      r += __shfl_xor(r, 1);
      r += __shfl_xor(r, 2);
      r += __shfl_xor(r, 4);
      r += __shfl_xor(r, 8);
      l_run[i] = l_run[i] * scale[i] + r;
    }
    #pragma unroll
    for (int t = 0; t < 8; ++t)
      #pragma unroll
      for (int i = 0; i < 8; ++i)
        acc[t][i] *= scale[i];

    // C-layout -> A-layout for P via LDS (dw==0 waves write; both halves read)
    if (dw == 0) {
      #pragma unroll
      for (int j = 0; j < 4; ++j)
        #pragma unroll
        for (int i = 0; i < 8; ++i)
          lds_st2(Pst + (mw * 16 + i + 8 * h) * PSTR + j * 16 + l16, f2bf(s[j][i]));
    }
    __syncthreads();

    // O += P V : K = 64 keys in two chunks of 32, all operands via ds_load
    #pragma unroll
    for (int kc = 0; kc < 2; ++kc) {
      const unsigned short* prow = Pst + (mw * 16 + l16) * PSTR + kc * 32;
      Frag fp;
      fp.q[0] = lds_ld16(prow + h * 8);
      fp.q[1] = lds_ld16(prow + 16 + h * 8);
      bf16x16 Ap = fp.v;
      #pragma unroll
      for (int t = 0; t < 8; ++t) {
        bf16x16 Bv = load_b_lds(Vs + (dw * 128 + t * 16) * VSTR + kc * 32, VSTR, lane);
        acc[t] = wmma_bf16(Ap, Bv, acc[t]);
      }
    }
  }

  // epilogue: out[b,s,k,d] = gamma * O[n][d]/l + x[b,s,k,d], n = k*64+s
  const float g = gamma[0];
  #pragma unroll
  for (int i = 0; i < 8; ++i) l_run[i] = 1.0f / l_run[i];
  #pragma unroll
  for (int t = 0; t < 8; ++t) {
    #pragma unroll
    for (int i = 0; i < 8; ++i) {
      int d = dw * 128 + t * 16 + l16;
      int n = qb + mw * 16 + i + 8 * h;
      int s_ = n & 63, k_ = n >> 6;
      size_t idx = (((size_t)b * 64 + s_) * 64 + k_) * DDIM + d;
      out[idx] = g * acc[t][i] * l_run[i] + x[idx];
    }
  }
}

extern "C" void kernel_launch(void* const* d_in, const int* in_sizes, int n_in,
                              void* d_out, int out_size, void* d_ws, size_t ws_size,
                              hipStream_t stream) {
  const float* x     = (const float*)d_in[0];
  const float* Wq    = (const float*)d_in[1];
  const float* bq    = (const float*)d_in[2];
  const float* Wk    = (const float*)d_in[3];
  const float* bk    = (const float*)d_in[4];
  const float* Wv    = (const float*)d_in[5];
  const float* bv    = (const float*)d_in[6];
  const float* gamma = (const float*)d_in[7];
  float* out = (float*)d_out;

  char* ws = (char*)d_ws;
  unsigned short* Xbf = (unsigned short*)(ws);             // 2*4096*256 bf16 = 4 MB
  unsigned short* Qbf = (unsigned short*)(ws + 4194304);   // 2*4096*32
  unsigned short* Kbf = (unsigned short*)(ws + 4718592);   // 2*4096*32
  unsigned short* Vt  = (unsigned short*)(ws + 5242880);   // 2*256*4096 = 4 MB
  unsigned short* Wqb = (unsigned short*)(ws + 9437184);   // 32*256
  unsigned short* Wkb = (unsigned short*)(ws + 9453568);   // 32*256
  unsigned short* Wvb = (unsigned short*)(ws + 9469952);   // 256*256

  const int totalX = 2 * NTOK * DDIM;
  k_convert_x<<<(totalX + 255) / 256, 256, 0, stream>>>(x, Xbf, totalX);
  k_convert_w<<<(81920 + 255) / 256, 256, 0, stream>>>(Wq, Wk, Wv, Wqb, Wkb, Wvb);
  k_proj_qk<<<dim3(NTOK / 128, 2), 256, 0, stream>>>(Xbf, Wqb, Wkb, bq, bk, Qbf, Kbf);
  k_proj_v<<<dim3(NTOK / 256, DDIM / 64, 2), 256, 0, stream>>>(Xbf, Wvb, bv, Vt);

  const size_t smem = (size_t)(2 * 256 * VSTR + 2 * 64 * KSTR + 64 * PSTR) * sizeof(unsigned short);
  k_attn<<<dim3(NTOK / 64, 2), 256, smem, stream>>>(Qbf, Kbf, Vt, x, gamma, out);
}
